// vGINEncoder_15556371546340
// MI455X (gfx1250) — matrix operand order repaired and checked by reading, not compiled, not run on hardware
//
#include <hip/hip_runtime.h>

// ---------------- problem constants ----------------
#define NN 100000   // nodes
#define NE 1600000  // edges
#define NG 512      // graphs
#define DH 128      // hidden / node dim
#define D2H 256     // 2*H

// ---------------- WMMA types ----------------
typedef _Float16 h8_t  __attribute__((ext_vector_type(8)));
typedef _Float16 v16h  __attribute__((ext_vector_type(16)));
typedef float    v8f   __attribute__((ext_vector_type(8)));
union AF16 { v16h v; h8_t h[2]; };

// ---------------- elementwise / utility kernels ----------------
__global__ void k_zero(float* __restrict__ p, int n) {
    int i = blockIdx.x * blockDim.x + threadIdx.x;
    if (i < n) p[i] = 0.f;
}

// o = (f16)(a + b)   -- fused GIN self+agg add (or pooled+vfeat) with f16 downcast for WMMA A
__global__ void k_cvt_add_f16(const float* __restrict__ a, const float* __restrict__ b,
                              _Float16* __restrict__ o, int n) {
    int i = blockIdx.x * blockDim.x + threadIdx.x;
    if (i < n) o[i] = (_Float16)(a[i] + b[i]);
}

// o = (f16)act(scale[c]*X + shift[c])  -- fused BN(+ReLU) + f16 downcast for WMMA A
__global__ void k_cvt_bn_f16(const float* __restrict__ X, const float* __restrict__ scale,
                             const float* __restrict__ shift, _Float16* __restrict__ o,
                             int n, int Ncols, int relu) {
    int i = blockIdx.x * blockDim.x + threadIdx.x;
    if (i >= n) return;
    int c = i % Ncols;
    float v = fmaf(X[i], scale[c], shift[c]);
    if (relu) v = fmaxf(v, 0.f);
    o[i] = (_Float16)v;
}

// post[node][f] += vfeat[batch[node]][f]   (in-place safe: 1 thread per element)
__global__ void k_add_vfeat(const float* __restrict__ post, const int* __restrict__ batch,
                            const float* __restrict__ vfeat, float* __restrict__ o, int n) {
    int i = blockIdx.x * blockDim.x + threadIdx.x;
    if (i >= n) return;
    int node = i >> 7, f = i & (DH - 1);
    o[i] = post[i] + vfeat[batch[node] * DH + f];
}

// agg[dst[e]] += X[src[e]]  — one wave (32 lanes) per edge, float4 per lane (L2-resident gather)
__global__ void k_edge_agg(const float* __restrict__ X, const int* __restrict__ src,
                           const int* __restrict__ dst, float* __restrict__ agg) {
    long long tid = (long long)blockIdx.x * blockDim.x + threadIdx.x;
    int e = (int)(tid >> 5);
    int lane = (int)(tid & 31);
    if (e >= NE) return;
    int s = src[e], d = dst[e];
    const float4 v = *(const float4*)&X[(size_t)s * DH + lane * 4];
    float* o = &agg[(size_t)d * DH + lane * 4];
    atomicAdd(o + 0, v.x); atomicAdd(o + 1, v.y);
    atomicAdd(o + 2, v.z); atomicAdd(o + 3, v.w);
}

// pooled[batch[n]] += post[n]  — one wave per node
__global__ void k_scatter_pool(const float* __restrict__ post, const int* __restrict__ batch,
                               float* __restrict__ pooled) {
    long long tid = (long long)blockIdx.x * blockDim.x + threadIdx.x;
    int nidx = (int)(tid >> 5);
    int lane = (int)(tid & 31);
    if (nidx >= NN) return;
    int g = batch[nidx];
    const float4 v = *(const float4*)&post[(size_t)nidx * DH + lane * 4];
    float* o = &pooled[(size_t)g * DH + lane * 4];
    atomicAdd(o + 0, v.x); atomicAdd(o + 1, v.y);
    atomicAdd(o + 2, v.z); atomicAdd(o + 3, v.w);
}

__global__ void k_counts(const int* __restrict__ batch, float* __restrict__ counts) {
    int i = blockIdx.x * blockDim.x + threadIdx.x;
    if (i < NN) atomicAdd(&counts[batch[i]], 1.0f);
}

__global__ void k_final(const float* __restrict__ pooled, const float* __restrict__ counts,
                        float* __restrict__ out) {
    int i = blockIdx.x * blockDim.x + threadIdx.x;
    if (i >= NG * DH) return;
    int g = i >> 7;
    out[i] = pooled[i] / fmaxf(counts[g], 1.0f);
}

__global__ void k_vfeat_init(const float* __restrict__ emb, float* __restrict__ vfeat) {
    int i = blockIdx.x * blockDim.x + threadIdx.x;
    if (i < NG * DH) vfeat[i] = emb[i & (DH - 1)];
}

// per-column sum / sum-of-squares (grid-strided over rows, atomic merge)
__global__ void k_colstats(const float* __restrict__ X, int M, int Ncols,
                           float* __restrict__ sums, float* __restrict__ sumsqs) {
    int tid = blockIdx.x * blockDim.x + threadIdx.x;
    int total = gridDim.x * blockDim.x;       // multiple of Ncols by construction
    int col = tid % Ncols;
    int r0 = tid / Ncols;
    int rs = total / Ncols;
    float s = 0.f, q = 0.f;
    for (int r = r0; r < M; r += rs) {
        float v = X[(size_t)r * Ncols + col];
        s += v; q += v * v;
    }
    atomicAdd(&sums[col], s);
    atomicAdd(&sumsqs[col], q);
}

// fold BN(mean, biased var, gamma, beta) into per-column scale/shift
__global__ void k_bn_finalize(const float* __restrict__ sums, const float* __restrict__ sumsqs,
                              const float* __restrict__ g, const float* __restrict__ b,
                              int M, float* __restrict__ scale, float* __restrict__ shift) {
    int c = threadIdx.x;
    float invM = 1.0f / (float)M;
    float mean = sums[c] * invM;
    float var = sumsqs[c] * invM - mean * mean;
    float sc = g[c] * rsqrtf(var + 1e-5f);
    scale[c] = sc;
    shift[c] = b[c] - mean * sc;
}

// O = act(scale[c]*X + shift[c])  (fp32 out — layer outputs that feed the edge gather)
__global__ void k_ewise_bn_act(const float* __restrict__ X, const float* __restrict__ scale,
                               const float* __restrict__ shift, float* __restrict__ O,
                               int n, int Ncols, int relu) {
    int i = blockIdx.x * blockDim.x + threadIdx.x;
    if (i >= n) return;
    int c = i % Ncols;
    float v = fmaf(X[i], scale[c], shift[c]);
    O[i] = relu ? fmaxf(v, 0.f) : v;
}

// ---------------- WMMA GEMM:  C = A(f16) @ W(f32->f16) + bias ----------------
// A: M x K row-major f16 (pre-converted, BN/add fused upstream); W: K x N fp32; C: M x N fp32.
// 128x128 block tile, BK=32 (one v_wmma_f32_16x16x32_f16 K-slab), double-buffered LDS with
// A staged via GLOBAL_LOAD_ASYNC_TO_LDS_B128 (ASYNCcnt) — no VGPR round-trip.
#define BM 128
#define BN 128
#define BK 32

__global__ __launch_bounds__(256) void k_gemm_wmma(
    const _Float16* __restrict__ A, const float* __restrict__ W,
    const float* __restrict__ bias, float* __restrict__ C,
    int M, int N, int K) {
    __shared__ _Float16 sA[2][BM * BK];   // [m][k]
    __shared__ _Float16 sB[2][BN * BK];   // [n][k] (transposed: lane K-runs contiguous)

    const int m0 = blockIdx.y * BM;
    const int n0 = blockIdx.x * BN;
    const int tid = threadIdx.x;
    const int lane = tid & 31;
    const int wave = tid >> 5;       // 8 waves
    const int waveM = wave & 1;      // 2 waves along M (64 rows)
    const int waveN = wave >> 1;     // 4 waves along N (32 cols)
    const int l16 = lane & 15;
    const int hi = lane >> 4;

    v8f acc[4][2];
#pragma unroll
    for (int i = 0; i < 4; i++)
#pragma unroll
        for (int j = 0; j < 2; j++)
#pragma unroll
            for (int e = 0; e < 8; e++) acc[i][j][e] = 0.f;

    const _Float16* Abase = A + (size_t)m0 * K;

    auto stage = [&](int k0, int buf) {
        // A tile: 512 x 16-byte chunks, async global(f16) -> LDS, layout-preserving.
        // Tail block rows read past M*K but stay inside the (over-allocated) buffer;
        // they only feed output rows >= M which are never stored.
#pragma unroll
        for (int t = 0; t < 2; t++) {
            int j = tid + t * 256;            // 0..511
            int r = j >> 2;                   // row in tile
            int c8 = (j & 3) * 8;             // half-offset in K-slab
            unsigned ldsa = (unsigned)(size_t)&sA[buf][r * BK + c8]; // flat addr[31:0] = LDS addr
            const void* g = (const void*)(Abase + (size_t)r * K + k0 + c8);
            asm volatile("global_load_async_to_lds_b128 %0, %1, off"
                         :: "v"(ldsa), "v"(g) : "memory");
        }
        // B tile: fp32 weights -> f16, stored transposed sB[n][k]; coalesced along N.
        for (int i = tid; i < BN * BK; i += 256) {
            int k = i >> 7, n = i & 127;
            int gn = n0 + n;
            float v = (gn < N) ? W[(size_t)(k0 + k) * N + gn] : 0.f;
            sB[buf][n * BK + k] = (_Float16)v;
        }
    };

    stage(0, 0);
    asm volatile("s_wait_asynccnt 0x0" ::: "memory");
    __syncthreads();

    const int nsteps = K >> 5;
    for (int s = 0; s < nsteps; s++) {
        const int cur = s & 1;
        if (s + 1 < nsteps) stage((s + 1) << 5, cur ^ 1);  // prefetch next slab into other buffer

        // B fragments: col = l16(+base); lanes 0-15: K 0-7 & 16-23; lanes 16-31: K 8-15 & 24-31
        AF16 bf[2];
#pragma unroll
        for (int nt = 0; nt < 2; nt++) {
            const h8_t* pB = (const h8_t*)&sB[cur][(waveN * 32 + nt * 16 + l16) * BK];
            bf[nt].h[0] = pB[hi];
            bf[nt].h[1] = pB[hi + 2];
        }
#pragma unroll
        for (int mt = 0; mt < 4; mt++) {
            // A fragment: row = l16(+base); lanes 0-15: K 0-7 & 16-23; lanes 16-31: K 8-15 & 24-31
            const h8_t* pA = (const h8_t*)&sA[cur][(waveM * 64 + mt * 16 + l16) * BK];
            AF16 af;
            af.h[0] = pA[hi];
            af.h[1] = pA[hi + 2];
#pragma unroll
            for (int nt = 0; nt < 2; nt++) {
                acc[mt][nt] = __builtin_amdgcn_wmma_f32_16x16x32_f16(
                    false, af.v, false, bf[nt].v, (short)0, acc[mt][nt], false, false);
            }
        }
        asm volatile("s_wait_asynccnt 0x0" ::: "memory");  // next slab landed in LDS
        __syncthreads();                                    // everyone done reading cur
    }

    // epilogue: D element (m = base + vgpr + 8*hi, n = base + l16), + bias
#pragma unroll
    for (int mt = 0; mt < 4; mt++)
#pragma unroll
        for (int nt = 0; nt < 2; nt++) {
            int gn = n0 + waveN * 32 + nt * 16 + l16;
            if (gn >= N) continue;
            float bv = bias ? bias[gn] : 0.f;
#pragma unroll
            for (int e = 0; e < 8; e++) {
                int gm = m0 + waveM * 64 + mt * 16 + hi * 8 + e;
                if (gm < M) C[(size_t)gm * N + gn] = acc[mt][nt][e] + bv;
            }
        }
}

// ---------------- host-side orchestration ----------------
static inline int cdiv(long long a, long long b) { return (int)((a + b - 1) / b); }

static void gemm(const _Float16* A, const float* W, const float* bias, float* C,
                 int M, int N, int K, hipStream_t s) {
    dim3 grid(cdiv(N, BN), cdiv(M, BM));
    k_gemm_wmma<<<grid, 256, 0, s>>>(A, W, bias, C, M, N, K);
}

// one GIN conv: agg -> Ah128=(f16)(in+agg) -> Lin1 -> BN stats -> Ah256=(f16)relu(bn(t1))
//               -> Lin2 -> outer BN stats -> post = act(...)
static void run_conv(const float* inX, const int* src, const int* dst,
                     const float* W1, const float* b1, const float* bng, const float* bnb,
                     const float* W2, const float* b2,
                     const float* outer_g, const float* outer_b, int relu_out,
                     float* agg, _Float16* Ah128, _Float16* Ah256,
                     float* t1, float* t2, float* post,
                     float* stats, float* scale, float* shift, hipStream_t s) {
    const int nH = NN * DH;
    k_zero<<<cdiv(nH, 256), 256, 0, s>>>(agg, nH);
    k_edge_agg<<<cdiv((long long)NE * 32, 256), 256, 0, s>>>(inX, src, dst, agg);
    k_cvt_add_f16<<<cdiv(nH, 256), 256, 0, s>>>(inX, agg, Ah128, nH);
    // Lin1: (NN x 128) @ (128 x 256)
    gemm(Ah128, W1, b1, t1, NN, D2H, DH, s);
    k_zero<<<2, 256, 0, s>>>(stats, 2 * D2H);
    k_colstats<<<512, 256, 0, s>>>(t1, NN, D2H, stats, stats + D2H);
    k_bn_finalize<<<1, D2H, 0, s>>>(stats, stats + D2H, bng, bnb, NN, scale, shift);
    // fused BN+ReLU+f16 downcast, then Lin2: (NN x 256) @ (256 x 128)
    k_cvt_bn_f16<<<cdiv(NN * D2H, 256), 256, 0, s>>>(t1, scale, shift, Ah256, NN * D2H, D2H, 1);
    gemm(Ah256, W2, b2, t2, NN, DH, D2H, s);
    k_zero<<<2, 256, 0, s>>>(stats, 2 * D2H);
    k_colstats<<<512, 256, 0, s>>>(t2, NN, DH, stats, stats + D2H);
    k_bn_finalize<<<1, DH, 0, s>>>(stats, stats + D2H, outer_g, outer_b, NN, scale, shift);
    k_ewise_bn_act<<<cdiv(nH, 256), 256, 0, s>>>(t2, scale, shift, post, nH, DH, relu_out);
}

extern "C" void kernel_launch(void* const* d_in, const int* in_sizes, int n_in,
                              void* d_out, int out_size, void* d_ws, size_t ws_size,
                              hipStream_t stream) {
    // ---- inputs (setup_inputs dict order) ----
    const float* x      = (const float*)d_in[0];
    const int*   ei     = (const int*)d_in[1];
    const int*   src    = ei;
    const int*   dst    = ei + NE;
    const int*   batch  = (const int*)d_in[2];
    const float* vn_emb = (const float*)d_in[3];
    const float* c1_W1 = (const float*)d_in[4];  const float* c1_b1 = (const float*)d_in[5];
    const float* c1_bng = (const float*)d_in[6]; const float* c1_bnb = (const float*)d_in[7];
    const float* c1_W2 = (const float*)d_in[8];  const float* c1_b2 = (const float*)d_in[9];
    const float* bn1_g = (const float*)d_in[10]; const float* bn1_b = (const float*)d_in[11];
    const float* cW1  = (const float*)d_in[12];  const float* cb1  = (const float*)d_in[13];
    const float* cbng = (const float*)d_in[14];  const float* cbnb = (const float*)d_in[15];
    const float* cW2  = (const float*)d_in[16];  const float* cb2  = (const float*)d_in[17];
    const float* bns_g = (const float*)d_in[18]; const float* bns_b = (const float*)d_in[19];
    const float* vW1 = (const float*)d_in[20];   const float* vb1 = (const float*)d_in[21];
    const float* vbn1_g = (const float*)d_in[22];const float* vbn1_b = (const float*)d_in[23];
    const float* vW2 = (const float*)d_in[24];   const float* vb2 = (const float*)d_in[25];
    const float* vbn2_g = (const float*)d_in[26];const float* vbn2_b = (const float*)d_in[27];
    float* out = (float*)d_out;

    // ---- workspace carve ----
    char* w = (char*)d_ws;
    size_t off = 0;
    auto alloc = [&](size_t bytes) -> char* {
        char* p = w + off;
        off = (off + bytes + 255) & ~(size_t)255;
        return p;
    };
    float* agg   = (float*)alloc((size_t)NN * DH * 4);
    float* t1    = (float*)alloc((size_t)NN * D2H * 4);
    float* t2    = (float*)alloc((size_t)NN * DH * 4);
    float* post  = (float*)alloc((size_t)NN * DH * 4);
    // f16 A operands, over-allocated by one 128-row block for async tail reads
    _Float16* Ah128 = (_Float16*)alloc((size_t)(NN + BM) * DH * 2);
    _Float16* Ah256 = (_Float16*)alloc((size_t)(NN + BM) * D2H * 2);
    float* stats = (float*)alloc(2 * D2H * 4);
    float* scale = (float*)alloc(D2H * 4);
    float* shift = (float*)alloc(D2H * 4);
    float* vfeat = (float*)alloc((size_t)NG * DH * 4);
    float* tv    = (float*)alloc((size_t)NG * D2H * 4);
    float* tv2   = (float*)alloc((size_t)NG * DH * 4);
    float* pooled= (float*)alloc((size_t)NG * DH * 4);
    float* counts= (float*)alloc(NG * 4);

    const int nH = NN * DH;
    const int gH = NG * DH;

    // vfeat = vn_emb broadcast
    k_vfeat_init<<<cdiv(gH, 256), 256, 0, stream>>>(vn_emb, vfeat);

    // ---- layer 1 : post = relu(bn1(gin_conv(x))) ----
    run_conv(x, src, dst, c1_W1, c1_b1, c1_bng, c1_bnb, c1_W2, c1_b2,
             bn1_g, bn1_b, /*relu*/1, agg, Ah128, Ah256, t1, t2, post,
             stats, scale, shift, stream);

    // ---- i = 0 ----
    k_add_vfeat<<<cdiv(nH, 256), 256, 0, stream>>>(post, batch, vfeat, post, nH);
    run_conv(post, src, dst,
             cW1 + 0 * DH * D2H, cb1 + 0 * D2H, cbng + 0 * D2H, cbnb + 0 * D2H,
             cW2 + 0 * D2H * DH, cb2 + 0 * DH, bns_g + 0 * DH, bns_b + 0 * DH,
             /*relu (i < L_EXTRA-1)*/1, agg, Ah128, Ah256, t1, t2, post,
             stats, scale, shift, stream);
    // virtual-node update
    k_zero<<<cdiv(gH, 256), 256, 0, stream>>>(pooled, gH);
    k_scatter_pool<<<cdiv((long long)NN * 32, 256), 256, 0, stream>>>(post, batch, pooled);
    k_cvt_add_f16<<<cdiv(gH, 256), 256, 0, stream>>>(pooled, vfeat, Ah128, gH);
    gemm(Ah128, vW1, vb1, tv, NG, D2H, DH, stream);
    k_zero<<<2, 256, 0, stream>>>(stats, 2 * D2H);
    k_colstats<<<512, 256, 0, stream>>>(tv, NG, D2H, stats, stats + D2H);
    k_bn_finalize<<<1, D2H, 0, stream>>>(stats, stats + D2H, vbn1_g, vbn1_b, NG, scale, shift);
    k_cvt_bn_f16<<<cdiv(NG * D2H, 256), 256, 0, stream>>>(tv, scale, shift, Ah256, NG * D2H, D2H, 1);
    gemm(Ah256, vW2, vb2, tv2, NG, DH, D2H, stream);
    k_zero<<<2, 256, 0, stream>>>(stats, 2 * D2H);
    k_colstats<<<512, 256, 0, stream>>>(tv2, NG, DH, stats, stats + D2H);
    k_bn_finalize<<<1, DH, 0, stream>>>(stats, stats + D2H, vbn2_g, vbn2_b, NG, scale, shift);
    k_ewise_bn_act<<<cdiv(gH, 256), 256, 0, stream>>>(tv2, scale, shift, vfeat, gH, DH, 1);

    // ---- i = 1 (no outer relu, no vnode update) ----
    k_add_vfeat<<<cdiv(nH, 256), 256, 0, stream>>>(post, batch, vfeat, post, nH);
    run_conv(post, src, dst,
             cW1 + 1 * DH * D2H, cb1 + 1 * D2H, cbng + 1 * D2H, cbnb + 1 * D2H,
             cW2 + 1 * D2H * DH, cb2 + 1 * DH, bns_g + 1 * DH, bns_b + 1 * DH,
             /*relu*/0, agg, Ah128, Ah256, t1, t2, post,
             stats, scale, shift, stream);

    // ---- readout: mean pool ----
    k_zero<<<cdiv(gH, 256), 256, 0, stream>>>(pooled, gH);
    k_scatter_pool<<<cdiv((long long)NN * 32, 256), 256, 0, stream>>>(post, batch, pooled);
    k_zero<<<cdiv(NG, 256), 256, 0, stream>>>(counts, NG);
    k_counts<<<cdiv(NN, 256), 256, 0, stream>>>(batch, counts);
    k_final<<<cdiv(gH, 256), 256, 0, stream>>>(pooled, counts, out);
}
